// RPN_12283606468110
// MI455X (gfx1250) — matrix operand order repaired and checked by reading, not compiled, not run on hardware
//
#include <hip/hip_runtime.h>
#include <hip/hip_bf16.h>

// ---------------------------------------------------------------------------
// RPN forward for MI455X (gfx1250, wave32, WMMA + TDM).
//   Stage 0: transpose features NCHW->NHWC bf16, weights OIHW->[tap][O][I] bf16
//   Stage 1: 3x3 conv as 9 shifted GEMMs via v_wmma_f32_16x16x32_bf16.
//            B (weight) tiles arrive via tensor_load_to_lds (TDM) into a
//            ping-pong LDS buffer, pipelined one chunk ahead and synchronized
//            with s_wait_tensorcnt. A tiles go VGPR->LDS (boundary zeroing).
//            Valid taps form a contiguous range [t0,t1) -> no tap table, and
//            the K-chunk loop fully unrolls (64 static WMMAs, no movrel).
//   Stage 2: cls/box heads + sigmoid + anchor decode + clip (VALU, LDS-tiled)
//   Stage 3: per-image radix-select top-2000, bitonic sort, sequential NMS,
//            rank-compact top-1000 (one 1024-thread WG per image, all in LDS)
// ---------------------------------------------------------------------------

typedef __bf16 v16bf __attribute__((ext_vector_type(16)));
typedef float  v8f   __attribute__((ext_vector_type(8)));
typedef unsigned int u32x4 __attribute__((ext_vector_type(4)));
typedef int          i32x4 __attribute__((ext_vector_type(4)));
typedef int          i32x8 __attribute__((ext_vector_type(8)));

union FragAB { v16bf v; unsigned int u[8]; };

#define BB      8
#define CC      256
#define HH      64
#define WW      64
#define AA      9
#define NANCH   (HH*WW*AA)        // 36864
#define NSEL    2000
#define NOUT    1000
#define NMS_T   0.7f
#define BCLIP   4.135166556742356f   // log(1000/16)

// LDS byte offset of a __shared__ object (ptrtoint of the AS(3) pointer).
typedef __attribute__((address_space(3))) unsigned char lds_byte_t;
static __device__ inline unsigned lds_offset_of(const void* p) {
    return (unsigned)(unsigned long long)(lds_byte_t*)p;
}

// Issue one TDM load: 2D tile, 256 rows x 32 bf16, row stride 256 elements,
// LDS rows padded to 72B (16 dwords data + 2 dwords pad) to match lBu[n*18].
static __device__ inline void tdm_load_btile(const __bf16* gsrc, unsigned lds_off) {
    const unsigned long long ga = (unsigned long long)gsrc;
    u32x4 g0;
    g0[0] = 1u;                                          // count=1, no gather
    g0[1] = lds_off;                                     // lds_addr (bytes)
    g0[2] = (unsigned)(ga & 0xFFFFFFFFull);              // global_addr[31:0]
    g0[3] = (unsigned)((ga >> 32) & 0x01FFFFFFull)       // global_addr[56:32]
          | (2u << 30);                                  // type = 2 (image)
    i32x8 g1;
    g1[0] = (1 << 16)      // data_size = 1 -> 2 bytes
          | (1 << 20)      // pad_enable
          | (3 << 22)      // pad_interval code 3 -> every 16 dwords (64B)
          | (1 << 25);     // pad_amount  code 1 -> 2 dwords (8B)
    const unsigned td0 = 1u << 20, td1 = 1u << 20;       // huge: no OOB clamp
    g1[1] = (int)((td0 & 0xFFFFu) << 16);                // tensor_dim0[15:0]
    g1[2] = (int)((td0 >> 16) | ((td1 & 0xFFFFu) << 16));// td0[31:16] | td1[15:0]
    g1[3] = (int)((td1 >> 16) | (32u << 16));            // td1[31:16] | tile_dim0=32
    g1[4] = 256;                                         // tile_dim1=256, tile_dim2=0
    g1[5] = 256;                                         // tensor_dim0_stride[31:0]
    g1[6] = 0;                                           // stride0[47:32] | stride1[15:0]
    g1[7] = 0;                                           // stride1[47:16]
    i32x4 gz = {0, 0, 0, 0};                             // 2-D tensor: groups 2/3 unused
#if defined(__clang_major__) && (__clang_major__ >= 23)
    i32x8 gz8 = {0, 0, 0, 0, 0, 0, 0, 0};
    __builtin_amdgcn_tensor_load_to_lds(g0, g1, gz, gz, gz8, 0);
#else
    __builtin_amdgcn_tensor_load_to_lds(g0, g1, gz, gz, 0);
#endif
}

// ---------------------------------------------------------------- stage 0 ---
__global__ void rpn_prep_feat(const float* __restrict__ feat, __bf16* __restrict__ featT) {
    size_t t = (size_t)blockIdx.x * blockDim.x + threadIdx.x;
    size_t total = (size_t)BB * HH * WW * CC;
    if (t >= total) return;
    unsigned c  = (unsigned)(t & 255u);
    unsigned p  = (unsigned)(t >> 8);       // b*4096 + yx
    unsigned b  = p >> 12;
    unsigned yx = p & 4095u;
    featT[t] = (__bf16)feat[((size_t)b * CC + c) * 4096 + yx];
}

__global__ void rpn_prep_wgt(const float* __restrict__ w, __bf16* __restrict__ wT) {
    size_t t = (size_t)blockIdx.x * blockDim.x + threadIdx.x;
    size_t total = (size_t)9 * CC * CC;
    if (t >= total) return;
    unsigned tap = (unsigned)(t >> 16);
    unsigned rem = (unsigned)(t & 65535u);
    unsigned co  = rem >> 8;
    unsigned ci  = rem & 255u;
    wT[t] = (__bf16)w[((size_t)co * CC + ci) * 9 + tap];
}

// ---------------------------------------------------------------- stage 1 ---
// One block = one (batch, row). Output tile: M=64 x-positions, N=256 channels.
// 8 waves; wave w owns N in [w*32, w*32+32), all 4 M-tiles -> 8 accum tiles.
__global__ __launch_bounds__(256) void rpn_conv3x3_wmma(
    const __bf16* __restrict__ feat,   // [B][64][64][256] bf16 (NHWC)
    const __bf16* __restrict__ wgt,    // [9][256(co)][256(ci)] bf16
    const float*  __restrict__ bias,
    float*        __restrict__ xout)   // [B][256][64][64] f32
{
    const int b    = blockIdx.x >> 6;
    const int y    = blockIdx.x & 63;
    const int tid  = threadIdx.x;
    const int wave = tid >> 5;
    const int lane = tid & 31;
    const int mrow = lane & 15;
    const int g    = lane >> 4;

    __shared__ unsigned int lAu[64 * 18];        // A tile  64x32 bf16, 36-half rows
    __shared__ unsigned int lBu[2][256 * 18];    // B tiles 256x32 bf16, ping-pong

    // valid taps are a contiguous range (rows clamp only top/bottom)
    const int t0 = (y > 0)  ? 0 : 3;
    const int t1 = (y < 63) ? 9 : 6;

    // prime the TDM pipeline: first chunk -> buffer 0 (wave 0 only)
    if (wave == 0)
        tdm_load_btile(wgt + (size_t)t0 * 65536, lds_offset_of(&lBu[0][0]));

    v8f acc[8];
    #pragma unroll
    for (int i = 0; i < 8; ++i)
        #pragma unroll
        for (int e = 0; e < 8; ++e) acc[i][e] = 0.f;

    int buf = 0;
    for (int tap = t0; tap < t1; ++tap) {
        const int dx = tap % 3 - 1;
        const int ys = y + tap / 3 - 1;
        const bool last_tap = (tap == t1 - 1);

        #pragma unroll
        for (int kcs = 0; kcs < 8; ++kcs) {
            const int kc = kcs * 32;

            __syncthreads();   // lAu reusable; lBu[buf^1] drained

            // --- A tile: 64 pixels x 32 channels, boundary-zeroed ----------
            {
                const int m   = tid >> 2;            // x position
                const int prt = tid & 3;
                const int xs  = m + dx;
                uint4 val = {0u, 0u, 0u, 0u};
                if (xs >= 0 && xs < WW) {
                    const uint4* src = reinterpret_cast<const uint4*>(
                        feat + ((((size_t)b * HH + ys) * WW + xs) * CC + kc));
                    val = src[prt];
                }
                unsigned int* dst = &lAu[m * 18 + prt * 4];
                dst[0] = val.x; dst[1] = val.y; dst[2] = val.z; dst[3] = val.w;
            }

            // --- TDM: prefetch next B tile, wait for the current one -------
            if (wave == 0) {
                if (kcs < 7) {
                    tdm_load_btile(wgt + ((size_t)tap * 65536 + kc + 32),
                                   lds_offset_of(&lBu[buf ^ 1][0]));
                    __builtin_amdgcn_s_wait_tensorcnt(1);
                } else if (!last_tap) {
                    tdm_load_btile(wgt + (size_t)(tap + 1) * 65536,
                                   lds_offset_of(&lBu[buf ^ 1][0]));
                    __builtin_amdgcn_s_wait_tensorcnt(1);
                } else {
                    __builtin_amdgcn_s_wait_tensorcnt(0);
                }
            }
            __syncthreads();

            // --- fragments + WMMA ------------------------------------------
            FragAB afr[4], bfr[2];
            #pragma unroll
            for (int mt = 0; mt < 4; ++mt) {
                const unsigned int* rowp = &lAu[(mt * 16 + mrow) * 18];
                #pragma unroll
                for (int j = 0; j < 8; ++j)
                    afr[mt].u[j] = rowp[(j & 3) + (j >> 2) * 8 + g * 4];
            }
            #pragma unroll
            for (int nt = 0; nt < 2; ++nt) {
                const unsigned int* rowp = &lBu[buf][((wave * 2 + nt) * 16 + mrow) * 18];
                #pragma unroll
                for (int j = 0; j < 8; ++j)
                    bfr[nt].u[j] = rowp[(j & 3) + (j >> 2) * 8 + g * 4];
            }
            #pragma unroll
            for (int mt = 0; mt < 4; ++mt)
                #pragma unroll
                for (int nt = 0; nt < 2; ++nt)
                    acc[mt * 2 + nt] = __builtin_amdgcn_wmma_f32_16x16x32_bf16(
                        false, afr[mt].v, false, bfr[nt].v,
                        (short)0, acc[mt * 2 + nt], false, false);
            buf ^= 1;
        }
    }

    // --- epilogue: bias + ReLU, store NCHW f32 -----------------------------
    #pragma unroll
    for (int mt = 0; mt < 4; ++mt)
        #pragma unroll
        for (int nt = 0; nt < 2; ++nt) {
            const int co = wave * 32 + nt * 16 + mrow;
            const float bs = bias[co];
            float* dst = xout + (((size_t)b * CC + co) * HH + y) * WW;
            #pragma unroll
            for (int i = 0; i < 8; ++i) {
                const int xp = mt * 16 + g * 8 + i;
                float v = acc[mt * 2 + nt][i] + bs;
                dst[xp] = v > 0.f ? v : 0.f;
            }
        }
}

// ---------------------------------------------------------------- stage 2 ---
__global__ __launch_bounds__(256) void rpn_heads_decode(
    const float* __restrict__ x,      // [B][256][64][64]
    const float* __restrict__ cls_w, const float* __restrict__ cls_b,
    const float* __restrict__ box_w, const float* __restrict__ box_b,
    const int*   __restrict__ imsz_p,
    float* __restrict__ obj,          // [B][36864]
    float* __restrict__ prop)         // [B][36864][4]
{
    const int b    = blockIdx.x >> 7;
    const int pc   = blockIdx.x & 127;
    const int pix0 = pc * 32;
    const int tid  = threadIdx.x;

    __shared__ float xs[CC][33];
    __shared__ float hout[32][48];

    {   // load 32 pixels x 256 channels (pixels contiguous in NCHW)
        const float* src = x + ((size_t)b * CC + tid) * 4096 + pix0;
        #pragma unroll
        for (int p = 0; p < 32; ++p) xs[tid][p] = src[p];
    }
    __syncthreads();

    for (int task = tid; task < 32 * 45; task += 256) {
        const int p = task / 45, o = task % 45;
        const float* w = (o < AA) ? (cls_w + o * CC) : (box_w + (o - AA) * CC);
        const float  bs = (o < AA) ? cls_b[o] : box_b[o - AA];
        float s = 0.f;
        #pragma unroll 8
        for (int c = 0; c < CC; ++c) s += xs[c][p] * w[c];
        hout[p][o] = s + bs;
    }
    __syncthreads();

    const float imsz   = (float)imsz_p[0];
    const float stride = imsz / 64.f;
    const float sizes[3]  = {128.f, 256.f, 512.f};
    const float ratios[3] = {0.5f, 1.f, 2.f};

    for (int task = tid; task < 32 * AA; task += 256) {
        const int p = task / AA, a = task % AA;
        const int pix = pix0 + p;
        const int yy = pix >> 6, xx = pix & 63;

        const float logit = hout[p][a];
        obj[(size_t)b * NANCH + (size_t)pix * AA + a] = 1.f / (1.f + __expf(-logit));

        const float dxv = hout[p][AA + a * 4 + 0];
        const float dyv = hout[p][AA + a * 4 + 1];
        const float dwv = fminf(hout[p][AA + a * 4 + 2], BCLIP);
        const float dhv = fminf(hout[p][AA + a * 4 + 3], BCLIP);

        const float sz = sizes[a / 3], sr = __fsqrt_rn(ratios[a % 3]);
        const float ha = sz * sr, wa = sz / sr;
        const float cxa = ((float)xx + 0.5f) * stride;
        const float cya = ((float)yy + 0.5f) * stride;

        const float cx = dxv * wa + cxa;
        const float cy = dyv * ha + cya;
        const float wb = __expf(dwv) * wa;
        const float hb = __expf(dhv) * ha;

        float4 bb;
        bb.x = fminf(fmaxf(cx - 0.5f * wb, 0.f), imsz);
        bb.y = fminf(fmaxf(cy - 0.5f * hb, 0.f), imsz);
        bb.z = fminf(fmaxf(cx + 0.5f * wb, 0.f), imsz);
        bb.w = fminf(fmaxf(cy + 0.5f * hb, 0.f), imsz);
        reinterpret_cast<float4*>(prop)[(size_t)b * NANCH + (size_t)pix * AA + a] = bb;
    }
}

// ---------------------------------------------------------------- stage 3 ---
__device__ inline unsigned long long rpn_key(float s, unsigned n) {
    unsigned u = __float_as_uint(s);
    u = (u & 0x80000000u) ? ~u : (u | 0x80000000u);          // monotonic map
    return ((unsigned long long)u << 32) | (unsigned)(~n);   // asc-idx tiebreak
}

__global__ __launch_bounds__(1024) void rpn_select_nms(
    const float* __restrict__ obj, const float* __restrict__ prop,
    float* __restrict__ out)          // [B][1000][4]
{
    const int b   = blockIdx.x;
    const int tid = threadIdx.x;
    const float* S = obj + (size_t)b * NANCH;
    const float4* P4 = reinterpret_cast<const float4*>(prop) + (size_t)b * NANCH;

    __shared__ unsigned int hist[256];
    __shared__ unsigned long long psh;
    __shared__ int ksh, cnt;
    __shared__ unsigned long long cand[2048];
    __shared__ float bxs[NSEL][4];
    __shared__ float ar[NSEL];
    __shared__ unsigned int keepw[64];
    __shared__ int wsum[65];

    // ---- MSB radix-select: exact 2000th-largest 64-bit key -----------------
    unsigned long long prefix = 0ull;
    int K = NSEL;
    for (int byte = 7; byte >= 0; --byte) {
        for (int i = tid; i < 256; i += 1024) hist[i] = 0u;
        __syncthreads();
        const int shift = byte * 8;
        const unsigned long long hm = (byte == 7) ? 0ull : (~0ull << (shift + 8));
        for (int n = tid; n < NANCH; n += 1024) {
            unsigned long long k = rpn_key(S[n], (unsigned)n);
            if ((k & hm) == prefix)
                atomicAdd(&hist[(unsigned)(k >> shift) & 255u], 1u);
        }
        __syncthreads();
        if (tid == 0) {
            int cum = 0, bsel = 0;
            for (int bin = 255; bin >= 0; --bin) {
                const int c = (int)hist[bin];
                if (cum + c >= K) { bsel = bin; break; }
                cum += c;
            }
            psh = prefix | ((unsigned long long)bsel << shift);
            ksh = K - cum;
        }
        __syncthreads();
        prefix = psh; K = ksh;
        __syncthreads();
    }
    const unsigned long long T = prefix;   // count(key >= T) == 2000 (keys distinct)

    // ---- compact candidates ------------------------------------------------
    if (tid == 0) cnt = 0;
    for (int i = tid; i < 2048; i += 1024) cand[i] = 0ull;
    __syncthreads();
    for (int n = tid; n < NANCH; n += 1024) {
        unsigned long long k = rpn_key(S[n], (unsigned)n);
        if (k >= T) {
            int p = atomicAdd(&cnt, 1);
            if (p < 2048) cand[p] = k;
        }
    }

    // ---- bitonic sort 2048 keys, descending (score desc, idx asc) ---------
    for (unsigned ksz = 2; ksz <= 2048; ksz <<= 1)
        for (unsigned j = ksz >> 1; j > 0; j >>= 1) {
            __syncthreads();
            for (unsigned i = tid; i < 2048; i += 1024) {
                const unsigned ixj = i ^ j;
                if (ixj > i) {
                    const unsigned long long a = cand[i], c2 = cand[ixj];
                    const bool desc = ((i & ksz) == 0);
                    if (desc ? (a < c2) : (a > c2)) { cand[i] = c2; cand[ixj] = a; }
                }
            }
        }
    __syncthreads();

    // ---- gather top-2000 boxes + areas ------------------------------------
    for (int i = tid; i < NSEL; i += 1024) {
        const unsigned idx = ~(unsigned)(cand[i] & 0xFFFFFFFFull);
        const float4 bb = P4[idx];
        bxs[i][0] = bb.x; bxs[i][1] = bb.y; bxs[i][2] = bb.z; bxs[i][3] = bb.w;
        ar[i] = (bb.z - bb.x) * (bb.w - bb.y);
    }
    for (int i = tid; i < 64; i += 1024)
        keepw[i] = (i < 62) ? 0xFFFFFFFFu : ((i == 62) ? 0x0000FFFFu : 0u);
    __syncthreads();

    // ---- sequential NMS (serial over i, parallel over j) -------------------
    for (int i = 0; i < NSEL - 1; ++i) {
        __syncthreads();
        if (!((keepw[i >> 5] >> (i & 31)) & 1u)) continue;   // uniform
        const float x0 = bxs[i][0], y0 = bxs[i][1], x1 = bxs[i][2], y1 = bxs[i][3];
        const float ai = ar[i];
        for (int j = i + 1 + tid; j < NSEL; j += 1024) {
            if (!((keepw[j >> 5] >> (j & 31)) & 1u)) continue;
            const float lx = fmaxf(x0, bxs[j][0]);
            const float ly = fmaxf(y0, bxs[j][1]);
            const float rx = fminf(x1, bxs[j][2]);
            const float ry = fminf(y1, bxs[j][3]);
            const float iw = fmaxf(rx - lx, 0.f);
            const float ih = fmaxf(ry - ly, 0.f);
            const float inter = iw * ih;
            const float iou = inter / (ai + ar[j] - inter + 1e-9f);
            if (iou > NMS_T) atomicAnd(&keepw[j >> 5], ~(1u << (j & 31)));
        }
    }
    __syncthreads();

    // ---- rank kept boxes, emit top-1000 (zero-fill tail) -------------------
    if (tid == 0) {
        int s = 0;
        for (int w2 = 0; w2 < 64; ++w2) { wsum[w2] = s; s += __popc(keepw[w2]); }
        wsum[64] = s;
    }
    __syncthreads();
    float* OUT = out + (size_t)b * NOUT * 4;
    for (int i = tid; i < NOUT * 4; i += 1024) OUT[i] = 0.f;
    __syncthreads();
    for (int i = tid; i < NSEL; i += 1024) {
        const unsigned wbits = keepw[i >> 5];
        if ((wbits >> (i & 31)) & 1u) {
            const int rank = wsum[i >> 5] + __popc(wbits & ((1u << (i & 31)) - 1u));
            if (rank < NOUT) {
                OUT[rank * 4 + 0] = bxs[i][0];
                OUT[rank * 4 + 1] = bxs[i][1];
                OUT[rank * 4 + 2] = bxs[i][2];
                OUT[rank * 4 + 3] = bxs[i][3];
            }
        }
    }
}

// ---------------------------------------------------------------- launch ----
extern "C" void kernel_launch(void* const* d_in, const int* in_sizes, int n_in,
                              void* d_out, int out_size, void* d_ws, size_t ws_size,
                              hipStream_t stream) {
    const float* features = (const float*)d_in[0];
    const float* conv_w   = (const float*)d_in[1];
    const float* conv_b   = (const float*)d_in[2];
    const float* cls_w    = (const float*)d_in[3];
    const float* cls_b    = (const float*)d_in[4];
    const float* box_w    = (const float*)d_in[5];
    const float* box_b    = (const float*)d_in[6];
    const int*   imsz     = (const int*)d_in[7];
    float* outp = (float*)d_out;

    char* ws = (char*)d_ws;
    size_t off = 0;
    auto take = [&](size_t bytes) {
        char* p = ws + off;
        off = (off + bytes + 255) & ~(size_t)255;
        return p;
    };
    __bf16* featT = (__bf16*)take((size_t)BB * HH * WW * CC * sizeof(__bf16));
    __bf16* wgtT  = (__bf16*)take((size_t)9 * CC * CC * sizeof(__bf16));
    float*  xact  = (float*)take((size_t)BB * CC * HH * WW * sizeof(float));
    float*  obj   = (float*)take((size_t)BB * NANCH * sizeof(float));
    float*  prop  = (float*)take((size_t)BB * NANCH * 4 * sizeof(float));

    {   // stage 0
        const size_t nf = (size_t)BB * HH * WW * CC;
        rpn_prep_feat<<<(unsigned)((nf + 255) / 256), 256, 0, stream>>>(features, featT);
        const size_t nw = (size_t)9 * CC * CC;
        rpn_prep_wgt<<<(unsigned)((nw + 255) / 256), 256, 0, stream>>>(conv_w, wgtT);
    }
    // stage 1: one block per (batch,row)
    rpn_conv3x3_wmma<<<BB * HH, 256, 0, stream>>>(featT, wgtT, conv_b, xact);
    // stage 2: one block per 32 pixels
    rpn_heads_decode<<<BB * 128, 256, 0, stream>>>(xact, cls_w, cls_b, box_w, box_b,
                                                   imsz, obj, prop);
    // stage 3: one block per image
    rpn_select_nms<<<BB, 1024, 0, stream>>>(obj, prop, outp);
}
